// QuantizedConv2d_12635793785612
// MI455X (gfx1250) — 8-bit matrix operand order repaired and checked by reading, compiled, not run on hardware
//
#include <hip/hip_runtime.h>
#include <hip/hip_bf16.h>
#include <stddef.h>


typedef __attribute__((ext_vector_type(8))) int v8i_t;
typedef int vec4i_t __attribute__((vector_size(16)));

#define CONV_B    8
#define CONV_CIN  320
#define CONV_COUT 320
#define CONV_H    64
#define CONV_W    64
#define PAD_H     66
#define PAD_W     66
#define XQ_PAD_BYTES ((size_t)CONV_B * PAD_H * PAD_W * CONV_CIN)

#if defined(__has_builtin)
#if __has_builtin(__builtin_amdgcn_global_load_async_to_lds_b128) && \
    __has_builtin(__builtin_amdgcn_s_wait_asynccnt)
#define USE_ASYNC_LDS 1
#endif
#endif
#ifndef USE_ASYNC_LDS
#define USE_ASYNC_LDS 0
#endif

__device__ __forceinline__ void ld_g2l_b128(const signed char* g, signed char* l) {
#if USE_ASYNC_LDS
    __builtin_amdgcn_global_load_async_to_lds_b128(
        (__attribute__((address_space(1))) vec4i_t*)g,
        (__attribute__((address_space(3))) vec4i_t*)l,
         0,  0);
#else
    *(int4*)l = *(const int4*)g;
#endif
}

__global__ __launch_bounds__(256)
void qconv_zero_kernel(int4* __restrict__ p, int n16) {
    const int i = blockIdx.x * 256 + threadIdx.x;
    if (i < n16) p[i] = make_int4(0, 0, 0, 0);
}

__global__ __launch_bounds__(256)
void qconv_quantize_kernel(const float* __restrict__ x,
                           const float* __restrict__ inv_p,
                           const float* __restrict__ zp_p,
                           signed char* __restrict__ xq) {
    const int idx = blockIdx.x * 256 + threadIdx.x;
    const int total = CONV_B * CONV_CIN * CONV_H * CONV_W;
    if (idx >= total) return;
    const float inv = inv_p[0];
    const float zp  = zp_p[0];
    const int xw = idx & (CONV_W - 1);
    const int yh = (idx >> 6) & (CONV_H - 1);
    const int c  = (idx >> 12) % CONV_CIN;
    const int b  = idx / (CONV_CIN * CONV_H * CONV_W);
    float v = rintf(x[idx] * inv) + zp;
    v = fminf(fmaxf(v, -128.0f), 127.0f);
    const size_t oidx =
        (((size_t)b * PAD_H + (yh + 1)) * PAD_W + (xw + 1)) * CONV_CIN + c;
    xq[oidx] = (signed char)v;
}

__global__ __launch_bounds__(256)
void qconv_packw_kernel(const int* __restrict__ w, signed char* __restrict__ wq) {
    const int idx = blockIdx.x * 256 + threadIdx.x;
    const int total = CONV_COUT * CONV_CIN * 9;
    if (idx >= total) return;
    const int kx = idx % 3;
    const int ky = (idx / 3) % 3;
    const int ci = (idx / 9) % CONV_CIN;
    const int o  = idx / (9 * CONV_CIN);
    wq[(((size_t)(ky * 3 + kx)) * CONV_COUT + o) * CONV_CIN + ci] = (signed char)w[idx];
}

__global__ __launch_bounds__(128)
void qconv_gemm_kernel(const signed char* __restrict__ xq,
                       const signed char* __restrict__ wq,
                       const float* __restrict__ wsum,
                       const float* __restrict__ scale,
                       const float* __restrict__ zp_p,
                       const float* __restrict__ bias,
                       float* __restrict__ out) {
    __shared__ __align__(16) signed char sA[2][64 * 64];
    __shared__ __align__(16) signed char sB[2][64 * 64];

    const int rowTile = blockIdx.x;
    const int b = rowTile >> 6;
    const int y = rowTile & (CONV_H - 1);
    const int obase = blockIdx.y * 64;

    const int t    = threadIdx.x;
    const int wave = t >> 5;
    const int lane = t & 31;
    const int lh   = lane >> 4;
    const int ll   = lane & 15;
    const int wm   = wave >> 1;
    const int wn   = wave & 1;

    const int ldRow  = t >> 1;
    const int ldHalf = t & 1;

    const signed char* xbase =
        xq + (((size_t)b * PAD_H + y) * PAD_W + ldRow) * CONV_CIN + ldHalf * 32;
    const signed char* wbase =
        wq + (size_t)(obase + ldRow) * CONV_CIN + ldHalf * 32;

    v8i_t acc[2][2];
#pragma unroll
    for (int i = 0; i < 2; ++i)
#pragma unroll
        for (int j = 0; j < 2; ++j) acc[i][j] = (v8i_t)0;

    auto issue = [&](int s, int buf) {
        const int tap = s / 5;
        const int cc  = s - tap * 5;
        const int tky = tap / 3;
        const int tkx = tap - tky * 3;
        const signed char* as = wbase + (size_t)tap * (CONV_COUT * CONV_CIN) + cc * 64;
        const signed char* bs = xbase + ((size_t)tky * PAD_W + tkx) * CONV_CIN + cc * 64;
        signed char* da = &sA[buf][ldRow * 64 + ldHalf * 32];
        signed char* db = &sB[buf][ldRow * 64 + ldHalf * 32];
        ld_g2l_b128(as,      da);
        ld_g2l_b128(as + 16, da + 16);
        ld_g2l_b128(bs,      db);
        ld_g2l_b128(bs + 16, db + 16);
    };

    auto compute = [&](int buf) {
        const signed char* a0p = &sA[buf][(wm * 32 + ll) * 64 + lh * 16];
        const signed char* a1p = a0p + 16 * 64;
        const int2 a00 = *(const int2*)(a0p + 0);
        const int2 a01 = *(const int2*)(a0p + 8);
        const int2 a02 = *(const int2*)(a0p + 32);
        const int2 a03 = *(const int2*)(a0p + 40);
        const v8i_t af0 = {a00.x, a00.y, a01.x, a01.y, a02.x, a02.y, a03.x, a03.y};
        const int2 a10 = *(const int2*)(a1p + 0);
        const int2 a11 = *(const int2*)(a1p + 8);
        const int2 a12 = *(const int2*)(a1p + 32);
        const int2 a13 = *(const int2*)(a1p + 40);
        const v8i_t af1 = {a10.x, a10.y, a11.x, a11.y, a12.x, a12.y, a13.x, a13.y};
#pragma unroll
        for (int j = 0; j < 2; ++j) {
            const signed char* bp = &sB[buf][(wn * 32 + j * 16 + ll) * 64 + lh * 16];
            const int4 r0 = *(const int4*)(bp + 0);
            const int4 r1 = *(const int4*)(bp + 32);
            const v8i_t bf = {r0.x, r0.y, r0.z, r0.w, r1.x, r1.y, r1.z, r1.w};
            acc[0][j] = __builtin_amdgcn_wmma_i32_16x16x64_iu8(
                true, af0, true, bf, acc[0][j], false, false);
            acc[1][j] = __builtin_amdgcn_wmma_i32_16x16x64_iu8(
                true, af1, true, bf, acc[1][j], false, false);
        }
    };

    issue(0, 0);
    for (int s = 0; s < 44; ++s) {
        issue(s + 1, (s + 1) & 1);
#if USE_ASYNC_LDS
        __builtin_amdgcn_s_wait_asynccnt(4);
#endif
        __syncthreads();
        compute(s & 1);
        __syncthreads();
    }
#if USE_ASYNC_LDS
    __builtin_amdgcn_s_wait_asynccnt(0);
#endif
    __syncthreads();
    compute(44 & 1);

    const float zp = zp_p[0];
#pragma unroll
    for (int i = 0; i < 2; ++i) {
#pragma unroll
        for (int j = 0; j < 2; ++j) {
            const int n = wn * 32 + j * 16 + ll;
#pragma unroll
            for (int r = 0; r < 8; ++r) {
                const int o = obase + wm * 32 + i * 16 + r + lh * 8;
                float corr = 0.0f;
#pragma unroll
                for (int ky2 = 0; ky2 < 3; ++ky2) {
                    const int yy2 = y + ky2 - 1;
                    if ((unsigned)yy2 >= (unsigned)CONV_H) continue;
#pragma unroll
                    for (int kx2 = 0; kx2 < 3; ++kx2) {
                        const int xx2 = n + kx2 - 1;
                        if ((unsigned)xx2 >= (unsigned)CONV_W) continue;
                        corr += wsum[o * 9 + ky2 * 3 + kx2];
                    }
                }
                const float val =
                    scale[o] * ((float)acc[i][j][r] - zp * corr) + bias[o];
                out[(((size_t)b * CONV_COUT + o) * CONV_H + y) * CONV_W + n] = val;
            }
        }
    }
}

extern "C" void kernel_launch(void* const* d_in, const int* in_sizes, int n_in,
                              void* d_out, int out_size, void* d_ws, size_t ws_size,
                              hipStream_t stream) {
    const float* x     = (const float*)d_in[0];
    const int*   wint  = (const int*)  d_in[1];
    const float* wsum  = (const float*)d_in[2];
    const float* scale = (const float*)d_in[3];
    const float* ainv  = (const float*)d_in[4];
    const float* azp   = (const float*)d_in[5];
    const float* bias  = (const float*)d_in[6];
    float* out = (float*)d_out;

    signed char* xq = (signed char*)d_ws;
    signed char* wq = xq + XQ_PAD_BYTES;

    {
        const int n16 = (int)(XQ_PAD_BYTES / 16);
        qconv_zero_kernel<<<(n16 + 255) / 256, 256, 0, stream>>>((int4*)xq, n16);
    }
    {
        const int total = CONV_B * CONV_CIN * CONV_H * CONV_W;
        qconv_quantize_kernel<<<(total + 255) / 256, 256, 0, stream>>>(x, ainv, azp, xq);
    }
    {
        const int total = CONV_COUT * CONV_CIN * 9;
        qconv_packw_kernel<<<(total + 255) / 256, 256, 0, stream>>>(wint, wq);
    }
    {
        dim3 grid(CONV_B * CONV_H, CONV_COUT / 64);
        qconv_gemm_kernel<<<grid, 128, 0, stream>>>(xq, wq, wsum, scale, azp, bias, out);
    }
}
